// DINOBevAlignerDeform_16990890623391
// MI455X (gfx1250) — matrix operand order repaired and checked by reading, not compiled, not run on hardware
//
#include <hip/hip_runtime.h>
#include <math.h>

#define NVIEW 6
#define HP 37
#define WPD 66
#define NTOK (HP * WPD)      // 2442
#define CD 768
#define CC 256
#define BEVH 100
#define BEVW 100
#define QN (BEVH * BEVW)     // 10000
#define NHD 8
#define NPT 4
#define HD (CC / NHD)        // 32
#define EPSF 1e-6f
#define SCALEF 0.5775f
#define W2M1 923.0f
#define H2M1 517.0f

typedef __attribute__((ext_vector_type(16))) __bf16 v16bf;
typedef __attribute__((ext_vector_type(8)))  float  v8f;

// ---- WMMA fragment helpers (CDNA5 16x16x32 bf16, wave32) ----
// A 16x32 (MxK) 16-bit layout (ISA 7.12.2): lane<16 -> row M=lane, K pairs
// {0..7, 16..23}; lane>=16 -> row M=lane-16, K pairs {8..15, 24..31}.
__device__ __forceinline__ v16bf load_a_frag_f32(const float* __restrict__ A, int M, int K,
                                                 int m0, int k0, int lane) {
  int row = m0 + (lane & 15);
  if (row >= M) row = M - 1;                       // clamp; stores are guarded
  const float* p = A + (size_t)row * K + k0 + ((lane & 16) ? 8 : 0);
  float4 f0 = *(const float4*)(p);
  float4 f1 = *(const float4*)(p + 4);
  float4 f2 = *(const float4*)(p + 16);
  float4 f3 = *(const float4*)(p + 20);
  v16bf a;
  a[0]  = (__bf16)f0.x; a[1]  = (__bf16)f0.y; a[2]  = (__bf16)f0.z; a[3]  = (__bf16)f0.w;
  a[4]  = (__bf16)f1.x; a[5]  = (__bf16)f1.y; a[6]  = (__bf16)f1.z; a[7]  = (__bf16)f1.w;
  a[8]  = (__bf16)f2.x; a[9]  = (__bf16)f2.y; a[10] = (__bf16)f2.z; a[11] = (__bf16)f2.w;
  a[12] = (__bf16)f3.x; a[13] = (__bf16)f3.y; a[14] = (__bf16)f3.z; a[15] = (__bf16)f3.w;
  return a;
}

// B 32x16 (KxN): lane holds column N=n0+(lane&15); K {0..7,16..23} (lane<16) or {8..15,24..31}.
__device__ __forceinline__ v16bf load_b_frag(const __bf16* __restrict__ B, int ldb,
                                             int k0, int n0, int lane) {
  int n = n0 + (lane & 15);
  int kb = k0 + ((lane & 16) ? 8 : 0);
  v16bf b;
#pragma unroll
  for (int s = 0; s < 8; ++s) {
    b[s]     = B[(size_t)(kb + s) * ldb + n];
    b[8 + s] = B[(size_t)(kb + 16 + s) * ldb + n];
  }
  return b;
}

// ---- 64-wide GEMM: wave computes 16x64 tile. A-fragment loaded once per K-step,
// ---- 4 independent WMMAs issued back-to-back (XDL co-exec friendly). Nt % 64 == 0.
__global__ void gemm_wmma_bf16_w64(const float* __restrict__ A, const __bf16* __restrict__ B,
                                   const float* __restrict__ bias, float* __restrict__ C,
                                   int M, int K, int Nt, int Mtiles) {
  int wid = blockIdx.x * (blockDim.x >> 5) + (threadIdx.x >> 5);
  int ntiles = Nt >> 6;
  int tm = wid / ntiles;
  if (tm >= Mtiles) return;                        // wave-uniform: EXEC stays all-ones
  int tn = wid - tm * ntiles;
  int lane = threadIdx.x & 31;
  int m0 = tm << 4, n0 = tn << 6;
  v8f acc[4];
#pragma unroll
  for (int j = 0; j < 4; ++j) acc[j] = (v8f){0.f, 0.f, 0.f, 0.f, 0.f, 0.f, 0.f, 0.f};
  for (int k0 = 0; k0 < K; k0 += 32) {
    if (k0 + 32 < K) {                             // gfx1250 global_prefetch_b8
      int row = m0 + (lane & 15); if (row >= M) row = M - 1;
      __builtin_prefetch(A + (size_t)row * K + k0 + 32, 0, 1);
    }
    v16bf a = load_a_frag_f32(A, M, K, m0, k0, lane);
#pragma unroll
    for (int j = 0; j < 4; ++j) {
      v16bf b = load_b_frag(B, Nt, k0, n0 + 16 * j, lane);
      acc[j] = __builtin_amdgcn_wmma_f32_16x16x32_bf16(false, a, false, b, (short)0,
                                                       acc[j], false, false);
    }
  }
  // C/D layout: lane<16 -> N=n0+lane, M=m0+r; lane>=16 -> N=n0+lane-16, M=m0+8+r
  int mb = m0 + ((lane & 16) ? 8 : 0);
#pragma unroll
  for (int j = 0; j < 4; ++j) {
    int n = n0 + 16 * j + (lane & 15);
    float bv = bias ? bias[n] : 0.0f;
#pragma unroll
    for (int r = 0; r < 8; ++r) {
      int m = mb + r;
      if (m < M) C[(size_t)m * Nt + n] = acc[j][r] + bv;
    }
  }
}

// ---- 16-wide generic GEMM (for small N, e.g. attention logits N=32) ----
__global__ void gemm_wmma_bf16(const float* __restrict__ A, const __bf16* __restrict__ B,
                               const float* __restrict__ bias, float* __restrict__ C,
                               int M, int K, int Nt, int Mtiles) {
  int wid = blockIdx.x * (blockDim.x >> 5) + (threadIdx.x >> 5);
  int ntiles = Nt >> 4;
  int tm = wid / ntiles;
  if (tm >= Mtiles) return;
  int tn = wid - tm * ntiles;
  int lane = threadIdx.x & 31;
  int m0 = tm << 4, n0 = tn << 4;
  v8f acc = {0.f, 0.f, 0.f, 0.f, 0.f, 0.f, 0.f, 0.f};
  for (int k0 = 0; k0 < K; k0 += 32) {
    v16bf a = load_a_frag_f32(A, M, K, m0, k0, lane);
    v16bf b = load_b_frag(B, Nt, k0, n0, lane);
    acc = __builtin_amdgcn_wmma_f32_16x16x32_bf16(false, a, false, b, (short)0, acc,
                                                  false, false);
  }
  int n = n0 + (lane & 15);
  int mb = m0 + ((lane & 16) ? 8 : 0);
  float bv = bias ? bias[n] : 0.0f;
#pragma unroll
  for (int r = 0; r < 8; ++r) {
    int m = mb + r;
    if (m < M) C[(size_t)m * Nt + n] = acc[r] + bv;
  }
}

// ---- final GEMM (16x64 tiles): out[c][q] = S[q,:]@opT[:,c] + op_b[c] + query[q,c] ----
__global__ void gemm_wmma_final(const float* __restrict__ S, const __bf16* __restrict__ opT,
                                const float* __restrict__ op_b, const float* __restrict__ query,
                                float* __restrict__ out) {
  int wid = blockIdx.x * (blockDim.x >> 5) + (threadIdx.x >> 5);
  const int ntiles = CC >> 6;                      // 4
  int tm = wid / ntiles;
  if (tm >= (QN >> 4)) return;
  int tn = wid - tm * ntiles;
  int lane = threadIdx.x & 31;
  int m0 = tm << 4, n0 = tn << 6;
  v8f acc[4];
#pragma unroll
  for (int j = 0; j < 4; ++j) acc[j] = (v8f){0.f, 0.f, 0.f, 0.f, 0.f, 0.f, 0.f, 0.f};
  for (int k0 = 0; k0 < CC; k0 += 32) {
    v16bf a = load_a_frag_f32(S, QN, CC, m0, k0, lane);
#pragma unroll
    for (int j = 0; j < 4; ++j) {
      v16bf b = load_b_frag(opT, CC, k0, n0 + 16 * j, lane);
      acc[j] = __builtin_amdgcn_wmma_f32_16x16x32_bf16(false, a, false, b, (short)0,
                                                       acc[j], false, false);
    }
  }
  int mb = m0 + ((lane & 16) ? 8 : 0);
#pragma unroll
  for (int j = 0; j < 4; ++j) {
    int n = n0 + 16 * j + (lane & 15);
    float bv = op_b[n];
#pragma unroll
    for (int r = 0; r < 8; ++r) {
      int m = mb + r;
      out[(size_t)n * QN + m] = acc[j][r] + bv + query[(size_t)m * CC + n];
    }
  }
}

// ---- elementwise fp32 -> bf16 ----
__global__ void cvt_bf16(const float* __restrict__ src, __bf16* __restrict__ dst, int n) {
  int i = blockIdx.x * blockDim.x + threadIdx.x;
  if (i < n) dst[i] = (__bf16)src[i];
}

// ---- transpose + cvt: src[R,C] fp32 -> dst[C,R] bf16 ----
__global__ void transpose_cvt_bf16(const float* __restrict__ src, __bf16* __restrict__ dst,
                                   int R, int C) {
  int i = blockIdx.x * blockDim.x + threadIdx.x;
  if (i >= R * C) return;
  int r = i / C, c = i - r * C;
  dst[(size_t)c * R + r] = (__bf16)src[i];
}

// ---- per-(q,head) softmax over NP=4 (in place) ----
__global__ void softmax_np(float* __restrict__ attn) {
  int t = blockIdx.x * blockDim.x + threadIdx.x;
  if (t >= QN * NHD) return;
  float* a = attn + (size_t)t * NPT;
  float m = fmaxf(fmaxf(a[0], a[1]), fmaxf(a[2], a[3]));
  float e0 = __expf(a[0] - m), e1 = __expf(a[1] - m);
  float e2 = __expf(a[2] - m), e3 = __expf(a[3] - m);
  float inv = 1.0f / (e0 + e1 + e2 + e3);
  a[0] = e0 * inv; a[1] = e1 * inv; a[2] = e2 * inv; a[3] = e3 * inv;
}

// ---- reference-point projection: ref2d[v][q][2] ----
__global__ void ref_points(const float* __restrict__ l2i, float* __restrict__ ref2d) {
  int t = blockIdx.x * blockDim.x + threadIdx.x;
  if (t >= NVIEW * QN) return;
  int v = t / QN, q = t - v * QN;
  const float* Mx = l2i + v * 16;
  int hy = q / BEVW, wx = q - hy * BEVW;
  float gx = (wx + 0.5f) * (1.0f / BEVW);
  float gy = (hy + 0.5f) * (1.0f / BEVH);
  float px = gx * 102.4f - 51.2f;
  float py = gy * 102.4f - 51.2f;
  float wsum = 0.f, rxs = 0.f, rys = 0.f;
#pragma unroll
  for (int d = 0; d < 4; ++d) {
    float zn = (0.5f + (float)d * (7.0f / 3.0f)) * 0.125f;
    float pz = zn * 8.0f - 5.0f;
    float c0 = Mx[0] * px + Mx[1] * py + Mx[2]  * pz + Mx[3];
    float c1 = Mx[4] * px + Mx[5] * py + Mx[6]  * pz + Mx[7];
    float c2 = Mx[8] * px + Mx[9] * py + Mx[10] * pz + Mx[11];
    bool bmask = c2 > 1e-5f;
    float dc = fmaxf(c2, 1e-5f);
    float u = (c0 / dc) * SCALEF;                  // PL = 0
    float vv = (c1 / dc) * SCALEF;                 // PT = 0
    bool in_img = (u >= 0.f) && (u <= W2M1) && (vv >= 0.f) && (vv <= H2M1);
    float w = (bmask && in_img) ? 1.0f : EPSF;
    float refx = fminf(fmaxf(u * (1.0f / W2M1), 0.f), 1.f);
    float refy = fminf(fmaxf(vv * (1.0f / H2M1), 0.f), 1.f);
    wsum += w; rxs += refx * w; rys += refy * w;
  }
  float inv = 1.0f / fmaxf(wsum, EPSF);
  ref2d[(size_t)t * 2 + 0] = rxs * inv;
  ref2d[(size_t)t * 2 + 1] = rys * inv;
}

// ---- deformable bilinear gather + view combine: S[q][c] = sum_v wn_v * samp_v[q][c] ----
__global__ void deform_sample(const float* __restrict__ val, const float* __restrict__ ref2d,
                              const float* __restrict__ off, const float* __restrict__ attn,
                              const float* __restrict__ w_view, float* __restrict__ S) {
  int t = blockIdx.x * blockDim.x + threadIdx.x;
  if (t >= QN * NHD) return;
  int q = t >> 3, h = t & 7;
  // normalized softplus view weights (sum to 1)
  float sp[NVIEW], ssum = 0.f;
#pragma unroll
  for (int v = 0; v < NVIEW; ++v) { sp[v] = log1pf(__expf(w_view[v])); ssum += sp[v]; }
  float sinv = 1.0f / fmaxf(ssum, EPSF);

  float4 acc[8];
#pragma unroll
  for (int i = 0; i < 8; ++i) acc[i] = make_float4(0.f, 0.f, 0.f, 0.f);

  const float* offp = off  + (size_t)q * (NHD * NPT * 2) + h * (NPT * 2);
  const float* attp = attn + (size_t)q * (NHD * NPT)     + h * NPT;

  for (int v = 0; v < NVIEW; ++v) {
    float rx = ref2d[((size_t)v * QN + q) * 2 + 0];
    float ry = ref2d[((size_t)v * QN + q) * 2 + 1];
    float wv = sp[v] * sinv;
    const float* vbase = val + (size_t)v * NTOK * CC + h * HD;
#pragma unroll
    for (int p = 0; p < NPT; ++p) {
      float aw = attp[p] * wv;
      float lx = rx + offp[2 * p + 0] * (1.0f / WPD);
      float ly = ry + offp[2 * p + 1] * (1.0f / HP);
      float x = lx * (float)WPD - 0.5f;
      float y = ly * (float)HP  - 0.5f;
      float x0f = floorf(x), y0f = floorf(y);
      float fx = x - x0f, fy = y - y0f;
      int x0 = (int)x0f, y0 = (int)y0f;
#pragma unroll
      for (int c = 0; c < 4; ++c) {
        int xi = x0 + (c & 1), yi = y0 + (c >> 1);
        float cw = ((c & 1) ? fx : 1.f - fx) * ((c >> 1) ? fy : 1.f - fy) * aw;
        if (xi < 0 || xi >= WPD || yi < 0 || yi >= HP) continue;
        const float4* g4 = (const float4*)(vbase + (size_t)(yi * WPD + xi) * CC);
#pragma unroll
        for (int i = 0; i < 8; ++i) {
          float4 g = g4[i];
          acc[i].x += cw * g.x; acc[i].y += cw * g.y;
          acc[i].z += cw * g.z; acc[i].w += cw * g.w;
        }
      }
    }
  }
  float4* s4 = (float4*)(S + (size_t)q * CC + h * HD);
#pragma unroll
  for (int i = 0; i < 8; ++i) s4[i] = acc[i];
}

extern "C" void kernel_launch(void* const* d_in, const int* in_sizes, int n_in,
                              void* d_out, int out_size, void* d_ws, size_t ws_size,
                              hipStream_t stream) {
  const float* last_tokens = (const float*)d_in[0];
  const float* lidar2img   = (const float*)d_in[1];
  const float* proj_w      = (const float*)d_in[2];
  const float* bev_query   = (const float*)d_in[3];
  const float* w_view      = (const float*)d_in[4];
  const float* vp_w        = (const float*)d_in[5];
  const float* vp_b        = (const float*)d_in[6];
  const float* so_w        = (const float*)d_in[7];
  const float* so_b        = (const float*)d_in[8];
  const float* aw_w        = (const float*)d_in[9];
  const float* aw_b        = (const float*)d_in[10];
  const float* op_w        = (const float*)d_in[11];
  const float* op_b        = (const float*)d_in[12];
  float* out = (float*)d_out;
  (void)in_sizes; (void)n_in; (void)out_size; (void)ws_size;

  char* ws = (char*)d_ws;
  size_t cur = 0;
  auto alloc = [&](size_t bytes) { size_t o = cur; cur += (bytes + 255) & ~(size_t)255; return o; };
  __bf16* proj_bf = (__bf16*)(ws + alloc((size_t)CC * CD * 2));   // proj_w as [K=256][N=768] bf16
  float*  Wf      = (float*) (ws + alloc((size_t)CC * CD * 4));   // fused vp_w @ proj_w [256,768]
  __bf16* WfT     = (__bf16*)(ws + alloc((size_t)CD * CC * 2));   // [768,256]
  __bf16* soT     = (__bf16*)(ws + alloc((size_t)CC * 64 * 2));   // [256,64]
  __bf16* awT     = (__bf16*)(ws + alloc((size_t)CC * 32 * 2));   // [256,32]
  __bf16* opT     = (__bf16*)(ws + alloc((size_t)CC * CC * 2));   // [256,256]
  float*  val     = (float*) (ws + alloc((size_t)NVIEW * NTOK * CC * 4));
  float*  offb    = (float*) (ws + alloc((size_t)QN * 64 * 4));
  float*  attnb   = (float*) (ws + alloc((size_t)QN * 32 * 4));
  float*  ref2d   = (float*) (ws + alloc((size_t)NVIEW * QN * 2 * 4));
  float*  Scomb   = (float*) (ws + alloc((size_t)QN * CC * 4));

  const int TB = 256;
  const int WPB = TB / 32;  // waves per block
  auto blocks_for_waves = [](int waves, int wpb) { return (waves + wpb - 1) / wpb; };

  // 1) proj_w fp32 -> bf16 (already [K,N] for the fusion GEMM)
  cvt_bf16<<<(CC * CD + TB - 1) / TB, TB, 0, stream>>>(proj_w, proj_bf, CC * CD);
  // 2) Wf = vp_w @ proj_w  (M=256, K=256, N=768)
  gemm_wmma_bf16_w64<<<blocks_for_waves((CC / 16) * (CD / 64), WPB), TB, 0, stream>>>(
      vp_w, proj_bf, nullptr, Wf, CC, CC, CD, CC / 16);
  // 3) transposed bf16 weights
  transpose_cvt_bf16<<<(CC * CD + TB - 1) / TB, TB, 0, stream>>>(Wf, WfT, CC, CD);
  transpose_cvt_bf16<<<(64 * CC + TB - 1) / TB, TB, 0, stream>>>(so_w, soT, 64, CC);
  transpose_cvt_bf16<<<(32 * CC + TB - 1) / TB, TB, 0, stream>>>(aw_w, awT, 32, CC);
  transpose_cvt_bf16<<<(CC * CC + TB - 1) / TB, TB, 0, stream>>>(op_w, opT, CC, CC);
  // 4) val = last_tokens @ WfT + vp_b  (M=14652, K=768, N=256)
  {
    int M = NVIEW * NTOK, Mtiles = (M + 15) / 16;
    gemm_wmma_bf16_w64<<<blocks_for_waves(Mtiles * (CC / 64), WPB), TB, 0, stream>>>(
        last_tokens, WfT, vp_b, val, M, CD, CC, Mtiles);
  }
  // 5) sampling offsets / attention logits from query (shared across views)
  gemm_wmma_bf16_w64<<<blocks_for_waves((QN / 16) * (64 / 64), WPB), TB, 0, stream>>>(
      bev_query, soT, so_b, offb, QN, CC, 64, QN / 16);
  gemm_wmma_bf16<<<blocks_for_waves((QN / 16) * (32 / 16), WPB), TB, 0, stream>>>(
      bev_query, awT, aw_b, attnb, QN, CC, 32, QN / 16);
  // 6) softmax over NP
  softmax_np<<<(QN * NHD + TB - 1) / TB, TB, 0, stream>>>(attnb);
  // 7) 3D -> 2D reference points per view
  ref_points<<<(NVIEW * QN + TB - 1) / TB, TB, 0, stream>>>(lidar2img, ref2d);
  // 8) bilinear gather + softplus view combine -> Scomb[Q,256]
  deform_sample<<<(QN * NHD + TB - 1) / TB, TB, 0, stream>>>(val, ref2d, offb, attnb,
                                                            w_view, Scomb);
  // 9) bev = Scomb @ opT + op_b + query, stored transposed as (CC, 100, 100)
  gemm_wmma_final<<<blocks_for_waves((QN / 16) * (CC / 64), WPB), TB, 0, stream>>>(
      Scomb, opT, op_b, bev_query, out);
}
